// EnhancedMSTSN_89928025244001
// MI455X (gfx1250) — matrix operand order repaired and checked by reading, not compiled, not run on hardware
//
#include <hip/hip_runtime.h>
#include <stdint.h>

typedef __attribute__((ext_vector_type(16))) _Float16 v16h;
typedef __attribute__((ext_vector_type(8)))  float    v8f;
typedef __attribute__((ext_vector_type(4)))  unsigned int u32x4;
typedef __attribute__((ext_vector_type(8)))  int      i32x8;
typedef __attribute__((ext_vector_type(4)))  int      i32x4;

// ---------------------------------------------------------------------------
// WMMA helpers for V_WMMA_F32_16X16X32_F16 (wave32).
// Layouts used by these helpers (A and B share a contiguous k order):
//   A (16x32 f16): lane l: m=l&15, g=l>>4; half hi -> K = 16*g + hi
//   B (32x16 f16): lane l: n=l&15, g=l>>4; half hi -> K = 16*g + hi
//   C/D (16x16 f32): vgpr vi -> row = 8*(l>>4)+vi, col = l&15
// ---------------------------------------------------------------------------
__device__ __forceinline__ v8f wmma_f16(v16h a, v16h b, v8f c) {
  return __builtin_amdgcn_wmma_f32_16x16x32_f16(false, a, false, b, (short)0, c,
                                                false, false);
}

__device__ __forceinline__ v16h frag_a_f16(const _Float16* base, int lda) {
  const int l = threadIdx.x & 31, m = l & 15, g = l >> 4;
  v16h a;
#pragma unroll
  for (int hi = 0; hi < 16; ++hi) {
    const int k = (g << 4) + hi;
    a[hi] = base[m * lda + k];
  }
  return a;
}

__device__ __forceinline__ v16h frag_a_f16_kmax(const _Float16* base, int lda, int kmax) {
  const int l = threadIdx.x & 31, m = l & 15, g = l >> 4;
  v16h a;
#pragma unroll
  for (int hi = 0; hi < 16; ++hi) {
    const int k = (g << 4) + hi;
    a[hi] = (k < kmax) ? base[m * lda + k] : (_Float16)0.f;
  }
  return a;
}

__device__ __forceinline__ v16h frag_b_f16(const _Float16* base, int ldb) {
  const int l = threadIdx.x & 31, n = l & 15, g = l >> 4;
  v16h b;
#pragma unroll
  for (int hi = 0; hi < 16; ++hi) b[hi] = base[((g << 4) + hi) * ldb + n];
  return b;
}

__device__ __forceinline__ v16h frag_b_f32(const float* base, int ldb) {
  const int l = threadIdx.x & 31, n = l & 15, g = l >> 4;
  v16h b;
#pragma unroll
  for (int hi = 0; hi < 16; ++hi) b[hi] = (_Float16)base[((g << 4) + hi) * ldb + n];
  return b;
}

// B[k][n] = src[n][k] (transposed source, f16 rows of length >= kmax)
__device__ __forceinline__ v16h frag_bT_f16(const _Float16* src, int ldn, int kmax) {
  const int l = threadIdx.x & 31, n = l & 15, g = l >> 4;
  v16h b;
#pragma unroll
  for (int hi = 0; hi < 16; ++hi) {
    const int k = (g << 4) + hi;
    b[hi] = (k < kmax) ? src[n * ldn + k] : (_Float16)0.f;
  }
  return b;
}

// B[k][n] = src[n][k] (transposed f32 source, full K=32)
__device__ __forceinline__ v16h frag_bT_f32(const float* src, int ldn) {
  const int l = threadIdx.x & 31, n = l & 15, g = l >> 4;
  v16h b;
#pragma unroll
  for (int hi = 0; hi < 16; ++hi) b[hi] = (_Float16)src[n * ldn + ((g << 4) + hi)];
  return b;
}

__device__ __forceinline__ void store_c_f16(_Float16* base, int ldc, v8f c, float biasc) {
  const int l = threadIdx.x & 31, col = l & 15, r0 = (l >> 4) * 8;
#pragma unroll
  for (int vi = 0; vi < 8; ++vi) base[(r0 + vi) * ldc + col] = (_Float16)(c[vi] + biasc);
}

__device__ __forceinline__ void store_c_f32(float* base, int ldc, v8f c, float biasc) {
  const int l = threadIdx.x & 31, col = l & 15, r0 = (l >> 4) * 8;
#pragma unroll
  for (int vi = 0; vi < 8; ++vi) base[(r0 + vi) * ldc + col] = c[vi] + biasc;
}

__device__ __forceinline__ void store_c_f32_scaled(float* base, int ldc, v8f c, float s) {
  const int l = threadIdx.x & 31, col = l & 15, r0 = (l >> 4) * 8;
#pragma unroll
  for (int vi = 0; vi < 8; ++vi) base[(r0 + vi) * ldc + col] = c[vi] * s;
}

__device__ __forceinline__ float gelu_f(float x) {
  return 0.5f * x * (1.0f + erff(x * 0.70710678118654752f));
}

// ===========================================================================
// Kernel 1: adjacency bitmask from cosine similarity of node embeddings.
// grid 256, block 256. maskb[i*8 + (j>>5)] bit (j&31).
// ===========================================================================
__global__ __launch_bounds__(256) void k_mask(const float* __restrict__ ne,
                                              uint32_t* __restrict__ maskb) {
  const int i = blockIdx.x, j = threadIdx.x;
  float dot = 0.f, ni = 0.f, nj = 0.f;
#pragma unroll
  for (int k = 0; k < 16; ++k) {
    const float a = ne[i * 16 + k], b = ne[j * 16 + k];
    dot += a * b; ni += a * a; nj += b * b;
  }
  const bool flag = dot > 0.5f * sqrtf(ni * nj);
  const unsigned long long bal = __ballot(flag);   // wave32: low 32 bits valid
  if ((j & 31) == 0) maskb[i * 8 + (j >> 5)] = (uint32_t)bal;
}

// ===========================================================================
// Kernel 2: fused projection + GAT1(relu) + GAT2 per graph (B*S = 384 graphs).
// block 256 (thread = node), online softmax over neighbors, LDS-resident.
// ===========================================================================
__global__ __launch_bounds__(256) void k_spatial(
    const float* __restrict__ inputs, const float* __restrict__ Wp,
    const float* __restrict__ bp, const float* __restrict__ W1g,
    const float* __restrict__ a1s, const float* __restrict__ a1d,
    const float* __restrict__ W2g, const float* __restrict__ a2s,
    const float* __restrict__ a2d, const uint32_t* __restrict__ maskb,
    float* __restrict__ spatial_out) {
  const int g = blockIdx.x;
  const int n = threadIdx.x;
  __shared__ __align__(16) char smem[57344];
  float* xp  = (float*)(smem + 0);       // [256][16], later reused as xp2
  float* h1  = (float*)(smem + 16384);   // [256][16]
  float* es1 = (float*)(smem + 32768);   // [256][4]
  float* ed1 = (float*)(smem + 36864);   // [256][4]
  float* h2  = (float*)(smem + 16384);   // [256][32] (reuses h1/es1/ed1)
  float* es2 = (float*)(smem + 0);       // [256][4]  (reuses xp2 region)
  float* ed2 = (float*)(smem + 4096);    // [256][4]
  uint32_t* mk = (uint32_t*)(smem + 49152); // [2048]

  for (int w = n; w < 2048; w += 256) mk[w] = maskb[w];

  // projection (row-local)
  const float xi0 = inputs[(g * 256 + n) * 3 + 0];
  const float xi1 = inputs[(g * 256 + n) * 3 + 1];
  const float xi2 = inputs[(g * 256 + n) * 3 + 2];
#pragma unroll
  for (int f = 0; f < 16; ++f)
    xp[n * 16 + f] = bp[f] + xi0 * Wp[f] + xi1 * Wp[16 + f] + xi2 * Wp[32 + f];

  // GAT1 features + attention logit halves (row-local)
  float hrow[16];
#pragma unroll
  for (int hd = 0; hd < 16; ++hd) {
    float a = 0.f;
#pragma unroll
    for (int f = 0; f < 16; ++f) a += xp[n * 16 + f] * W1g[f * 16 + hd];
    hrow[hd] = a;
    h1[n * 16 + hd] = a;
  }
#pragma unroll
  for (int h = 0; h < 4; ++h) {
    float s = 0.f, d = 0.f;
#pragma unroll
    for (int dd = 0; dd < 4; ++dd) {
      s += hrow[h * 4 + dd] * a1s[h * 4 + dd];
      d += hrow[h * 4 + dd] * a1d[h * 4 + dd];
    }
    es1[n * 4 + h] = s;
    ed1[n * 4 + h] = d;
  }
  __syncthreads();

  // GAT1 aggregation: online softmax over masked neighbors (D=4, H=4)
  {
    float m[4], ssum[4], acc[16], edr[4];
#pragma unroll
    for (int h = 0; h < 4; ++h) { m[h] = -1e30f; ssum[h] = 0.f; edr[h] = ed1[n * 4 + h]; }
#pragma unroll
    for (int q = 0; q < 16; ++q) acc[q] = 0.f;
    for (int j = 0; j < 256; ++j) {
      if (!((mk[n * 8 + (j >> 5)] >> (j & 31)) & 1u)) continue;
      const float* hj = h1 + j * 16;
#pragma unroll
      for (int h = 0; h < 4; ++h) {
        float l = edr[h] + es1[j * 4 + h];
        l = l > 0.f ? l : 0.2f * l;
        if (l > m[h]) {
          const float r = __expf(m[h] - l);
          ssum[h] *= r;
#pragma unroll
          for (int dd = 0; dd < 4; ++dd) acc[h * 4 + dd] *= r;
          m[h] = l;
        }
        const float p = __expf(l - m[h]);
        ssum[h] += p;
#pragma unroll
        for (int dd = 0; dd < 4; ++dd) acc[h * 4 + dd] += p * hj[h * 4 + dd];
      }
    }
    float* xp2 = xp;  // safe: region 0 unread during aggregation
#pragma unroll
    for (int hd = 0; hd < 16; ++hd) {
      const float v = acc[hd] / ssum[hd >> 2];
      xp2[n * 16 + hd] = v > 0.f ? v : 0.f;   // relu
    }
  }
  __syncthreads();

  // GAT2 features (row-local)
  float h2row[32];
#pragma unroll
  for (int c = 0; c < 32; ++c) {
    float a = 0.f;
#pragma unroll
    for (int f = 0; f < 16; ++f) a += xp[n * 16 + f] * W2g[f * 32 + c];
    h2row[c] = a;
    h2[n * 32 + c] = a;
  }
  __syncthreads();   // all xp2 reads done before es2/ed2 overwrite region 0
#pragma unroll
  for (int h = 0; h < 4; ++h) {
    float s = 0.f, d = 0.f;
#pragma unroll
    for (int dd = 0; dd < 8; ++dd) {
      s += h2row[h * 8 + dd] * a2s[h * 8 + dd];
      d += h2row[h * 8 + dd] * a2d[h * 8 + dd];
    }
    es2[n * 4 + h] = s;
    ed2[n * 4 + h] = d;
  }
  __syncthreads();

  // GAT2 aggregation (D=8)
  {
    float m[4], ssum[4], acc[32], edr[4];
#pragma unroll
    for (int h = 0; h < 4; ++h) { m[h] = -1e30f; ssum[h] = 0.f; edr[h] = ed2[n * 4 + h]; }
#pragma unroll
    for (int q = 0; q < 32; ++q) acc[q] = 0.f;
    for (int j = 0; j < 256; ++j) {
      if (!((mk[n * 8 + (j >> 5)] >> (j & 31)) & 1u)) continue;
      const float* hj = h2 + j * 32;
#pragma unroll
      for (int h = 0; h < 4; ++h) {
        float l = edr[h] + es2[j * 4 + h];
        l = l > 0.f ? l : 0.2f * l;
        if (l > m[h]) {
          const float r = __expf(m[h] - l);
          ssum[h] *= r;
#pragma unroll
          for (int dd = 0; dd < 8; ++dd) acc[h * 8 + dd] *= r;
          m[h] = l;
        }
        const float p = __expf(l - m[h]);
        ssum[h] += p;
#pragma unroll
        for (int dd = 0; dd < 8; ++dd) acc[h * 8 + dd] += p * hj[h * 8 + dd];
      }
    }
#pragma unroll
    for (int c = 0; c < 32; ++c)
      spatial_out[(g * 256 + n) * 32 + c] = acc[c] / ssum[c >> 3];
  }
}

// ===========================================================================
// Kernel 3: spatial_feats[b,n,:] = mean over S of spatial_out[b,s,n,:]
// ===========================================================================
__global__ __launch_bounds__(32) void k_smean(const float* __restrict__ spat,
                                              float* __restrict__ sfeat) {
  const int bn = blockIdx.x;            // b*256+n
  const int b = bn >> 8, n = bn & 255, c = threadIdx.x;
  float s = 0.f;
  for (int t = 0; t < 48; ++t) s += spat[((b * 48 + t) * 256 + n) * 32 + c];
  sfeat[bn * 32 + c] = s * (1.f / 48.f);
}

// ===========================================================================
// Kernel 4: temporal transformer layer per (b,n) sequence (S=48, dm=32),
// fully LDS-resident, WMMA everywhere. grid 2048, block 96 (3 waves = 3
// 16-row M-tiles). The strided [48][32] f32 sequence tile is fetched by the
// Tensor Data Mover (2D D#: tile 32x48, row stride 8192 elems) into LDS,
// tracked by TENSORcnt. Writes only the S-mean (temporal_feats).
// ===========================================================================
__global__ __launch_bounds__(96) void k_temporal(
    const float* __restrict__ spatial_out,
    const float* __restrict__ Wq, const float* __restrict__ bq,
    const float* __restrict__ Wk, const float* __restrict__ bk,
    const float* __restrict__ Wv, const float* __restrict__ bv,
    const float* __restrict__ Wo, const float* __restrict__ bo,
    const float* __restrict__ W1f, const float* __restrict__ b1f,
    const float* __restrict__ W2f, const float* __restrict__ b2f,
    const float* __restrict__ g1, const float* __restrict__ be1,
    const float* __restrict__ g2, const float* __restrict__ be2,
    float* __restrict__ temporal_feats) {
  const int bn = blockIdx.x;
  const int b = bn >> 8, n = bn & 255;
  const int tid = threadIdx.x;
  const int wave = tid >> 5;
  const int lane = tid & 31;
  const int col = lane & 15;
  const int r0 = (lane >> 4) * 8;

  __builtin_prefetch(Wq, 0, 3);
  __builtin_prefetch(W1f, 0, 3);

  __shared__ __align__(16) char smem[59392];
  _Float16* tf    = (_Float16*)(smem + 0);      // [48][32]  input (f16)
  _Float16* qs    = (_Float16*)(smem + 3072);   // [2][48][16]
  _Float16* ks    = (_Float16*)(smem + 6144);   // [2][48][16]
  _Float16* vs    = (_Float16*)(smem + 9216);   // [2][64][16] K-padded
  float*    sc    = (float*)(smem + 13312);     // [2][48][48] scores
  float*    stage = (float*)(smem + 13312);     // [48][32] f32 TDM staging (pre-sc)
  _Float16* hidh  = (_Float16*)(smem + 13312);  // [48][64]   (reuses sc)
  _Float16* pm    = (_Float16*)(smem + 31744);  // [2][48][64] probs K-padded
  _Float16* oh    = (_Float16*)(smem + 31744);  // [48][32]   (reuses pm)
  float*    ob    = (float*)(smem + 44032);     // [48][32]  attn output
  float*    out1  = (float*)(smem + 50176);     // [48][32]
  _Float16* out1h = (_Float16*)(smem + 56320);  // [48][32]

#if __has_builtin(__builtin_amdgcn_tensor_load_to_lds) && __has_builtin(__builtin_amdgcn_s_wait_tensorcnt)
  // ---- TDM: DMA the strided [48][32] f32 tile into LDS (ISA ch.7 TDM) ----
  if (wave == 0) {
    const uint64_t ga = (uint64_t)(uintptr_t)(spatial_out +
                        ((uint64_t)(b * 48) * 256 + (uint64_t)n) * 32);
    const uint32_t lds_off = (uint32_t)(uintptr_t)stage;  // low 32 bits = LDS byte offset
    u32x4 g0;
    g0[0] = 1u;                                            // count=1 valid descriptor
    g0[1] = lds_off;                                       // lds_addr
    g0[2] = (uint32_t)ga;                                  // global_addr[31:0]
    g0[3] = (uint32_t)((ga >> 32) & 0x01FFFFFFu) | (2u << 30); // addr[56:32] | type=2
    i32x8 g1d;
    g1d[0] = 2 << 16;          // data_size = 2 (4 bytes), no mask/flags
    g1d[1] = 32 << 16;         // tensor_dim0 = 32 (low 16 at bits 63:48)
    g1d[2] = 48 << 16;         // tensor_dim0 hi=0 | tensor_dim1 = 48
    g1d[3] = 32 << 16;         // tensor_dim1 hi=0 | tile_dim0 = 32
    g1d[4] = 48;               // tile_dim1 = 48, tile_dim2 = 0 (2D)
    g1d[5] = 8192;             // tensor_dim0_stride = 8192 elements (N*32)
    g1d[6] = 0;                // stride hi | tensor_dim1_stride lo
    g1d[7] = 0;
    i32x4 gz = {0, 0, 0, 0};
#if defined(__clang_major__) && (__clang_major__ >= 23)
    i32x8 gz8 = {0, 0, 0, 0, 0, 0, 0, 0};
    __builtin_amdgcn_tensor_load_to_lds(g0, g1d, gz, gz, gz8, 0);
#else
    __builtin_amdgcn_tensor_load_to_lds(g0, g1d, gz, gz, 0);
#endif
    __builtin_amdgcn_s_wait_tensorcnt(0);
  }
  __syncthreads();
  for (int idx = tid; idx < 48 * 32; idx += 96) tf[idx] = (_Float16)stage[idx];
#else
  // Fallback: plain vector loads
  for (int idx = tid; idx < 48 * 32; idx += 96) {
    const int s = idx >> 5, c = idx & 31;
    tf[idx] = (_Float16)spatial_out[((b * 48 + s) * 256 + n) * 32 + c];
  }
#endif
  for (int idx = tid; idx < 2 * 64 * 16; idx += 96) vs[idx] = (_Float16)0.f;
  __syncthreads();

  // ---- QKV projections: [16x32] x [32x16] per head ----
  {
    const v16h at = frag_a_f16(tf + wave * 16 * 32, 32);
#pragma unroll
    for (int h = 0; h < 2; ++h) {
      v8f cq = {}; cq = wmma_f16(at, frag_b_f32(Wq + h * 16, 32), cq);
      store_c_f16(qs + h * 768 + wave * 256, 16, cq, bq[h * 16 + col]);
      v8f ck = {}; ck = wmma_f16(at, frag_b_f32(Wk + h * 16, 32), ck);
      store_c_f16(ks + h * 768 + wave * 256, 16, ck, bk[h * 16 + col]);
      v8f cv = {}; cv = wmma_f16(at, frag_b_f32(Wv + h * 16, 32), cv);
      store_c_f16(vs + h * 1024 + wave * 256, 16, cv, bv[h * 16 + col]);
    }
  }
  __syncthreads();

  // ---- scores = Q K^T / sqrt(16) (head dim 16 padded to K=32) ----
#pragma unroll
  for (int h = 0; h < 2; ++h) {
    const v16h aq = frag_a_f16_kmax(qs + h * 768 + wave * 256, 16, 16);
#pragma unroll
    for (int jt = 0; jt < 3; ++jt) {
      v8f c = {};
      c = wmma_f16(aq, frag_bT_f16(ks + h * 768 + jt * 256, 16, 16), c);
      store_c_f32_scaled(sc + h * 2304 + wave * 16 * 48 + jt * 16, 48, c, 0.25f);
    }
  }
  __syncthreads();

  // ---- softmax over 48 keys, write f16 probs padded to 64 ----
  {
    const int h = tid / 48, i = tid % 48;  // 96 threads == 2*48 rows
    const float* row = sc + h * 2304 + i * 48;
    float mx = row[0];
    for (int j = 1; j < 48; ++j) mx = fmaxf(mx, row[j]);
    float ssum = 0.f;
    for (int j = 0; j < 48; ++j) ssum += __expf(row[j] - mx);
    const float ri = 1.f / ssum;
    _Float16* prow = pm + h * 3072 + i * 64;
    for (int j = 0; j < 48; ++j) prow[j] = (_Float16)(__expf(row[j] - mx) * ri);
    for (int j = 48; j < 64; ++j) prow[j] = (_Float16)0.f;
  }
  __syncthreads();

  // ---- attn @ V: K = 48 padded to 64, 2 chunks ----
#pragma unroll
  for (int h = 0; h < 2; ++h) {
    v8f c = {};
#pragma unroll
    for (int kc = 0; kc < 2; ++kc) {
      c = wmma_f16(frag_a_f16(pm + h * 3072 + wave * 16 * 64 + kc * 32, 64),
                   frag_b_f16(vs + h * 1024 + kc * 512, 16), c);
    }
    store_c_f32(ob + wave * 16 * 32 + h * 16, 32, c, 0.f);
  }
  __syncthreads();

  for (int idx = tid; idx < 1536; idx += 96) oh[idx] = (_Float16)ob[idx];  // reuse pm
  __syncthreads();

  // ---- Wo projection + bias + residual ----
  {
    const v16h ao = frag_a_f16(oh + wave * 16 * 32, 32);
#pragma unroll
    for (int nt = 0; nt < 2; ++nt) {
      v8f c = {};
      c = wmma_f16(ao, frag_b_f32(Wo + nt * 16, 32), c);
      const int cc = nt * 16 + col;
      const float bb = bo[cc];
#pragma unroll
      for (int vi = 0; vi < 8; ++vi) {
        const int r = wave * 16 + r0 + vi;
        out1[r * 32 + cc] = c[vi] + bb + (float)tf[r * 32 + cc];
      }
    }
  }
  __syncthreads();

  // ---- LayerNorm 1 ----
  if (tid < 48) {
    float* row = out1 + tid * 32;
    float mu = 0.f;
    for (int c = 0; c < 32; ++c) mu += row[c];
    mu *= (1.f / 32.f);
    float var = 0.f;
    for (int c = 0; c < 32; ++c) { const float d = row[c] - mu; var += d * d; }
    var *= (1.f / 32.f);
    const float rs = rsqrtf(var + 1e-3f);
    _Float16* rh = out1h + tid * 32;
    for (int c = 0; c < 32; ++c) {
      const float yv = (row[c] - mu) * rs * g1[c] + be1[c];
      row[c] = yv;
      rh[c] = (_Float16)yv;
    }
  }
  __syncthreads();

  // ---- FFN dense1 (32->64) + exact GELU ----
  {
    const v16h a1 = frag_a_f16(out1h + wave * 16 * 32, 32);
#pragma unroll
    for (int nt = 0; nt < 4; ++nt) {
      v8f c = {};
      c = wmma_f16(a1, frag_b_f32(W1f + nt * 16, 64), c);
      const int cc = nt * 16 + col;
      const float bb = b1f[cc];
#pragma unroll
      for (int vi = 0; vi < 8; ++vi) {
        const int r = wave * 16 + r0 + vi;
        hidh[r * 64 + cc] = (_Float16)gelu_f(c[vi] + bb);
      }
    }
  }
  __syncthreads();

  // ---- FFN dense2 (64->32) + residual ----
#pragma unroll
  for (int nt = 0; nt < 2; ++nt) {
    v8f c = {};
#pragma unroll
    for (int kc = 0; kc < 2; ++kc) {
      c = wmma_f16(frag_a_f16(hidh + wave * 16 * 64 + kc * 32, 64),
                   frag_b_f32(W2f + kc * 32 * 32 + nt * 16, 32), c);
    }
    const int cc = nt * 16 + col;
    const float bb = b2f[cc];
#pragma unroll
    for (int vi = 0; vi < 8; ++vi) {
      const int r = wave * 16 + r0 + vi;
      out1[r * 32 + cc] = c[vi] + bb + out1[r * 32 + cc];
    }
  }
  __syncthreads();

  // ---- LayerNorm 2 (in place) ----
  if (tid < 48) {
    float* row = out1 + tid * 32;
    float mu = 0.f;
    for (int c = 0; c < 32; ++c) mu += row[c];
    mu *= (1.f / 32.f);
    float var = 0.f;
    for (int c = 0; c < 32; ++c) { const float d = row[c] - mu; var += d * d; }
    var *= (1.f / 32.f);
    const float rs = rsqrtf(var + 1e-3f);
    for (int c = 0; c < 32; ++c) row[c] = (row[c] - mu) * rs * g2[c] + be2[c];
  }
  __syncthreads();

  // ---- mean over S ----
  if (tid < 32) {
    float s = 0.f;
    for (int t = 0; t < 48; ++t) s += out1[t * 32 + tid];
    temporal_feats[(b * 256 + n) * 32 + tid] = s * (1.f / 48.f);
  }
}

// ===========================================================================
// Kernel 5: cross-attention QKV projections (2048 rows, K=32, N=64)
// ===========================================================================
__global__ __launch_bounds__(64) void k_cqkv(
    const float* __restrict__ sfeat, const float* __restrict__ tfeat,
    const float* __restrict__ Wq, const float* __restrict__ bq,
    const float* __restrict__ Wk, const float* __restrict__ bk,
    const float* __restrict__ Wv, const float* __restrict__ bv,
    float* __restrict__ qc, float* __restrict__ kc, float* __restrict__ vc) {
  const int row = blockIdx.x, j = threadIdx.x;
  const float* xs = sfeat + row * 32;
  const float* xt = tfeat + row * 32;
  float aq = bq[j], ak = bk[j], av = bv[j];
#pragma unroll
  for (int d = 0; d < 32; ++d) {
    aq += xs[d] * Wq[d * 64 + j];
    ak += xt[d] * Wk[d * 64 + j];
    av += xt[d] * Wv[d * 64 + j];
  }
  qc[row * 64 + j] = aq;
  kc[row * 64 + j] = ak;
  vc[row * 64 + j] = av;
}

// ===========================================================================
// Kernel 6: cross attention core. grid = 8b * 16 qtiles * 2 heads = 256,
// one wave per block: 16 WMMAs for QK^T (K=32), softmax over 256,
// 8x2 WMMAs for P@V.
// ===========================================================================
__global__ __launch_bounds__(32) void k_cattn(const float* __restrict__ qc,
                                              const float* __restrict__ kc,
                                              const float* __restrict__ vc,
                                              float* __restrict__ oc) {
  const int blk = blockIdx.x;
  const int b = blk >> 5;
  const int it = (blk >> 1) & 15;
  const int h = blk & 1;
  const int tid = threadIdx.x;
  const int col = tid & 15, r0 = (tid >> 4) * 8;

  __shared__ __align__(16) char smem[25600];
  float* sc = (float*)smem;                   // [16][256]
  _Float16* pm = (_Float16*)(smem + 16384);   // [16][256]
  _Float16* qt = (_Float16*)(smem + 24576);   // [16][32]

  for (int idx = tid; idx < 512; idx += 32) {
    const int r = idx >> 5, c = idx & 31;
    qt[idx] = (_Float16)qc[(b * 256 + it * 16 + r) * 64 + h * 32 + c];
  }
  __syncthreads();

  const v16h aq = frag_a_f16(qt, 32);
  for (int jt = 0; jt < 16; ++jt) {
    v8f c = {};
    c = wmma_f16(aq, frag_bT_f32(kc + (b * 256 + jt * 16) * 64 + h * 32, 64), c);
    store_c_f32_scaled(sc + jt * 16, 256, c, 0.176776695296637f);  // 1/sqrt(32)
  }
  __syncthreads();

  if (tid < 16) {
    float* row = sc + tid * 256;
    float mx = row[0];
    for (int j = 1; j < 256; ++j) mx = fmaxf(mx, row[j]);
    float ss = 0.f;
    for (int j = 0; j < 256; ++j) ss += __expf(row[j] - mx);
    const float ri = 1.f / ss;
    _Float16* prow = pm + tid * 256;
    for (int j = 0; j < 256; ++j) prow[j] = (_Float16)(__expf(row[j] - mx) * ri);
  }
  __syncthreads();

#pragma unroll
  for (int nt = 0; nt < 2; ++nt) {
    v8f c = {};
    for (int kcn = 0; kcn < 8; ++kcn) {
      c = wmma_f16(frag_a_f16(pm + kcn * 32, 256),
                   frag_b_f32(vc + (b * 256 + kcn * 32) * 64 + h * 32 + nt * 16, 64), c);
    }
#pragma unroll
    for (int vi = 0; vi < 8; ++vi) {
      const int r = it * 16 + r0 + vi;
      oc[(b * 256 + r) * 64 + h * 32 + nt * 16 + col] = c[vi];
    }
  }
}

// ===========================================================================
// Kernel 7: output projection + readout MLP (exact GELU) -> out[8,256]
// ===========================================================================
__global__ __launch_bounds__(128) void k_readout(
    const float* __restrict__ oc, const float* __restrict__ Wo,
    const float* __restrict__ bo, const float* __restrict__ Wr1,
    const float* __restrict__ br1, const float* __restrict__ Wr2,
    const float* __restrict__ br2, float* __restrict__ out) {
  const int row = blockIdx.x * 128 + threadIdx.x;
  if (row >= 2048) return;
  const float* orow = oc + row * 64;
  float fused[32];
#pragma unroll
  for (int d = 0; d < 32; ++d) fused[d] = bo[d];
  for (int k = 0; k < 64; ++k) {
    const float ov = orow[k];
#pragma unroll
    for (int d = 0; d < 32; ++d) fused[d] += ov * Wo[k * 32 + d];
  }
  float acc = br2[0];
#pragma unroll
  for (int j = 0; j < 16; ++j) {
    float r = br1[j];
#pragma unroll
    for (int d = 0; d < 32; ++d) r += fused[d] * Wr1[d * 16 + j];
    acc += gelu_f(r) * Wr2[j];
  }
  out[row] = acc;
}

// ===========================================================================
extern "C" void kernel_launch(void* const* d_in, const int* in_sizes, int n_in,
                              void* d_out, int out_size, void* d_ws, size_t ws_size,
                              hipStream_t stream) {
  (void)in_sizes; (void)n_in; (void)out_size; (void)ws_size;
  const float* inputs     = (const float*)d_in[0];
  const float* node_embed = (const float*)d_in[1];
  const float* Wp  = (const float*)d_in[2];
  const float* bp  = (const float*)d_in[3];
  const float* W1g = (const float*)d_in[4];
  const float* a1s = (const float*)d_in[5];
  const float* a1d = (const float*)d_in[6];
  const float* W2g = (const float*)d_in[7];
  const float* a2s = (const float*)d_in[8];
  const float* a2d = (const float*)d_in[9];
  const float* Wq_t = (const float*)d_in[10];
  const float* bq_t = (const float*)d_in[11];
  const float* Wk_t = (const float*)d_in[12];
  const float* bk_t = (const float*)d_in[13];
  const float* Wv_t = (const float*)d_in[14];
  const float* bv_t = (const float*)d_in[15];
  const float* Wo_t = (const float*)d_in[16];
  const float* bo_t = (const float*)d_in[17];
  const float* W1f = (const float*)d_in[18];
  const float* b1f = (const float*)d_in[19];
  const float* W2f = (const float*)d_in[20];
  const float* b2f = (const float*)d_in[21];
  const float* g1  = (const float*)d_in[22];
  const float* be1 = (const float*)d_in[23];
  const float* g2  = (const float*)d_in[24];
  const float* be2 = (const float*)d_in[25];
  const float* Wq_c = (const float*)d_in[26];
  const float* bq_c = (const float*)d_in[27];
  const float* Wk_c = (const float*)d_in[28];
  const float* bk_c = (const float*)d_in[29];
  const float* Wv_c = (const float*)d_in[30];
  const float* bv_c = (const float*)d_in[31];
  const float* Wo_c = (const float*)d_in[32];
  const float* bo_c = (const float*)d_in[33];
  const float* Wr1 = (const float*)d_in[34];
  const float* br1 = (const float*)d_in[35];
  const float* Wr2 = (const float*)d_in[36];
  const float* br2 = (const float*)d_in[37];

  char* ws = (char*)d_ws;
  uint32_t* maskb = (uint32_t*)(ws + 0);                  // 8 KB
  float* spat  = (float*)(ws + 8192);                     // 384*256*32 f32
  float* sfeat = (float*)(ws + 8192 + 12582912);          // 8*256*32
  float* tfeat = sfeat + 65536;
  float* qc = tfeat + 65536;
  float* kc = qc + 131072;
  float* vc = kc + 131072;
  float* oc = vc + 131072;

  k_mask<<<256, 256, 0, stream>>>(node_embed, maskb);
  k_spatial<<<384, 256, 0, stream>>>(inputs, Wp, bp, W1g, a1s, a1d, W2g, a2s, a2d,
                                     maskb, spat);
  k_smean<<<2048, 32, 0, stream>>>(spat, sfeat);
  k_temporal<<<2048, 96, 0, stream>>>(spat, Wq_t, bq_t, Wk_t, bk_t, Wv_t, bv_t,
                                      Wo_t, bo_t, W1f, b1f, W2f, b2f, g1, be1,
                                      g2, be2, tfeat);
  k_cqkv<<<2048, 64, 0, stream>>>(sfeat, tfeat, Wq_c, bq_c, Wk_c, bk_c, Wv_c, bv_c,
                                  qc, kc, vc);
  k_cattn<<<256, 32, 0, stream>>>(qc, kc, vc, oc);
  k_readout<<<16, 128, 0, stream>>>(oc, Wo_c, bo_c, Wr1, br1, Wr2, br2,
                                    (float*)d_out);
}
